// NewLLM2Layer_12893491822870
// MI455X (gfx1250) — compile-verified, matrix operand order checked by reading, not run on hardware
//
#include <hip/hip_runtime.h>

#define DEV __device__ __forceinline__

typedef float  v8f   __attribute__((ext_vector_type(8)));
typedef __bf16 v16bf __attribute__((ext_vector_type(16)));
typedef unsigned int u32x4 __attribute__((ext_vector_type(4)));
typedef int i32x4 __attribute__((ext_vector_type(4)));
typedef int i32x8 __attribute__((ext_vector_type(8)));

constexpr int V_ = 32000, E_ = 512, C_ = 512, H_ = 1024, B_ = 8, S_ = 512;
constexpr int EC_ = E_ + C_;   // 1024  (fnn_in width)
constexpr int HC_ = H_ + C_;   // 1536  (comb width)
constexpr int MP  = 16;        // padded batch rows for 16x16 WMMA tiles
constexpr int TG  = 4;         // timesteps per logits block (B-operand reuse)
// LDS row stride for TDM-padded A tiles: 2048B data + 2x16B pad per row
constexpr int LROW = 2080;

// ---------------- WMMA helpers (CDNA5 16x16x32 bf16) ----------------

union FragU { v16bf v; u32x4 q[2]; };

DEV v16bf make_frag(const __bf16* p0, const __bf16* p1) {
  FragU f;
  f.q[0] = *reinterpret_cast<const u32x4*>(p0);
  f.q[1] = *reinterpret_cast<const u32x4*>(p1);
  return f.v;
}

// Row-major [rows][ld] source. For A: row0=0, lane&15 = M. For B we store
// weights transposed [N][K], so lane&15 = N and the K pattern matches A.
// Per ISA 7.12.2: lanes<16 hold K{k0..k0+7, k0+16..k0+23}, lanes>=16 hold +8.
DEV v16bf load_rm_frag(const __bf16* base, int ld, int row0, int k0, int lane) {
  int r  = row0 + (lane & 15);
  int kk = k0 + ((lane & 16) ? 8 : 0);
  const __bf16* p = base + (size_t)r * ld + kk;
  return make_frag(p, p + 16);
}

// fnn_in[m][k] = k<512 ? tok_t[m][k] : ctx[m][k-512]   (chunks never straddle 512)
DEV v16bf load_fnn_frag(const __bf16* tok_t, const __bf16* ctxb, int lane, int k0) {
  int m  = lane & 15;
  int kk = k0 + ((lane & 16) ? 8 : 0);
  int k2 = kk + 16;
  const __bf16* p0 = (kk < E_) ? (tok_t + m * E_ + kk) : (ctxb + m * C_ + (kk - E_));
  const __bf16* p1 = (k2 < E_) ? (tok_t + m * E_ + k2) : (ctxb + m * C_ + (k2 - E_));
  return make_frag(p0, p1);
}

// comb[m][k] = k<1024 ? h2[m][k] : ctx[m][k-1024]
DEV v16bf load_comb_frag(const __bf16* h2b, const __bf16* ctxb, int lane, int k0) {
  int m  = lane & 15;
  int kk = k0 + ((lane & 16) ? 8 : 0);
  int k2 = kk + 16;
  const __bf16* p0 = (kk < H_) ? (h2b + m * H_ + kk) : (ctxb + m * C_ + (kk - H_));
  const __bf16* p1 = (k2 < H_) ? (h2b + m * H_ + k2) : (ctxb + m * C_ + (k2 - H_));
  return make_frag(p0, p1);
}

// A-frag from TDM-staged LDS tile (row stride LROW, +16B after col 512)
DEV v16bf load_lds_a_frag(const char* As, int row0, int lane, int k0) {
  int m  = row0 + (lane & 15);
  int kk = k0 + ((lane & 16) ? 8 : 0);
  int k2 = kk + 16;
  const char* row = As + (size_t)m * LROW;
  const __bf16* p0 = (const __bf16*)(row + kk * 2 + ((kk >= 512) ? 16 : 0));
  const __bf16* p1 = (const __bf16*)(row + k2 * 2 + ((k2 >= 512) ? 16 : 0));
  return make_frag(p0, p1);
}

DEV v8f wmma_bf16(v16bf a, v16bf b, v8f c) {
  return __builtin_amdgcn_wmma_f32_16x16x32_bf16(false, a, false, b, (short)0, c,
                                                 false, false);
}

// 8 waves each hold a partial 16x16 tile (K-split); reduce through LDS.
DEV float block_reduce_tile(float (*red)[256], const v8f& acc, int wave, int lane, int tid) {
  #pragma unroll
  for (int r = 0; r < 8; ++r) red[wave][r * 32 + lane] = acc[r];
  __syncthreads();
  float s = 0.f;
  #pragma unroll
  for (int w = 0; w < 8; ++w) s += red[w][tid];
  return s;
}

// ---------------- Tensor Data Mover: 2D bf16 tile -> padded LDS ----------------
// rows x 1024 bf16, global row stride 1024, LDS pad: +16B per 1024B (ISA 8.3-8.4)
DEV void tdm_load_tile(const __bf16* gsrc, unsigned lds_byte_addr, int rows) {
  unsigned long long ga = (unsigned long long)(uintptr_t)gsrc;
  u32x4 g0;
  g0[0] = 1u;                                        // count=1, user mode
  g0[1] = lds_byte_addr;                             // D#.lds_addr
  g0[2] = (unsigned)(ga & 0xFFFFFFFFu);              // global_addr[31:0]
  g0[3] = (unsigned)((ga >> 32) & 0x01FFFFFFu) | (2u << 30);  // addr[56:32] | type=2
  i32x8 g1;
  g1[0] = (1 << 16)          // data_size = 1 (2 bytes)
        | (1 << 20)          // pad_enable
        | (7 << 22)          // pad_interval: 256 DWORDs (1024B)
        | (3 << 25);         // pad_amount: 4 DWORDs (16B)
  g1[1] = (H_ & 0xFFFF) << 16;                 // tensor_dim0 lo = 1024
  g1[2] = (rows & 0xFFFF) << 16;               // tensor_dim0 hi=0 | tensor_dim1 lo
  g1[3] = (H_ & 0xFFFF) << 16;                 // tensor_dim1 hi=0 | tile_dim0 = 1024
  g1[4] = rows & 0xFFFF;                       // tile_dim1 = rows, tile_dim2 = 0
  g1[5] = H_;                                  // tensor_dim0_stride = 1024
  g1[6] = 0;
  g1[7] = 0;
  i32x4 z4 = {0, 0, 0, 0};
#if __clang_major__ >= 23
  i32x8 z8 = {0, 0, 0, 0, 0, 0, 0, 0};
  __builtin_amdgcn_tensor_load_to_lds(g0, g1, z4, z4, z8, 0);
#else
  __builtin_amdgcn_tensor_load_to_lds(g0, g1, z4, z4, 0);
#endif
}

// ---------------- device-wide sense-reversing barrier ----------------

DEV void grid_barrier(unsigned* counter, unsigned* sense, unsigned nb, unsigned& lsense) {
  __syncthreads();
  if (threadIdx.x == 0) {
    lsense ^= 1u;
    __threadfence();
    unsigned arrived = __hip_atomic_fetch_add(counter, 1u, __ATOMIC_ACQ_REL,
                                              __HIP_MEMORY_SCOPE_AGENT);
    if (arrived == nb - 1u) {
      __hip_atomic_store(counter, 0u, __ATOMIC_RELAXED, __HIP_MEMORY_SCOPE_AGENT);
      __hip_atomic_store(sense, lsense, __ATOMIC_RELEASE, __HIP_MEMORY_SCOPE_AGENT);
    } else {
      while (__hip_atomic_load(sense, __ATOMIC_ACQUIRE, __HIP_MEMORY_SCOPE_AGENT) != lsense) {
        __builtin_amdgcn_s_sleep(2);
      }
    }
  }
  __syncthreads();
}

// ---------------- phase 0: weight convert + transpose to bf16 ----------------
__global__ __launch_bounds__(256) void wt_transpose_kernel(const float* __restrict__ in,
                                                           __bf16* __restrict__ out,
                                                           int K, int N) {
  size_t total = (size_t)K * N;
  size_t stride = (size_t)gridDim.x * blockDim.x;
  for (size_t i = (size_t)blockIdx.x * blockDim.x + threadIdx.x; i < total; i += stride) {
    int n = (int)(i / K);
    int k = (int)(i % K);
    out[i] = (__bf16)in[(size_t)k * N + n];
  }
}

// ---------------- phase 1: embedding gather + LayerNorm -> bf16 [S][16][E] ----------------
__global__ __launch_bounds__(256) void embed_ln_kernel(const int* __restrict__ ids,
                                                       const float* __restrict__ emb,
                                                       const float* __restrict__ g,
                                                       const float* __restrict__ bb,
                                                       __bf16* __restrict__ tok) {
  int bs = blockIdx.x;
  int b  = bs / S_;
  int s  = bs % S_;
  int id = ids[b * S_ + s];
  const float* row = emb + (size_t)id * E_;
  int tid = threadIdx.x, lane = tid & 31, wave = tid >> 5;

  float x0 = row[tid], x1 = row[tid + 256];
  float ls = x0 + x1, lq = x0 * x0 + x1 * x1;
  #pragma unroll
  for (int off = 16; off > 0; off >>= 1) {
    ls += __shfl_xor(ls, off, 32);
    lq += __shfl_xor(lq, off, 32);
  }
  __shared__ float ssum[8], ssq[8], bc[2];
  if (lane == 0) { ssum[wave] = ls; ssq[wave] = lq; }
  __syncthreads();
  if (tid == 0) {
    float a = 0.f, q = 0.f;
    #pragma unroll
    for (int w = 0; w < 8; ++w) { a += ssum[w]; q += ssq[w]; }
    float mean = a * (1.f / E_);
    float var  = q * (1.f / E_) - mean * mean;
    bc[0] = mean;
    bc[1] = rsqrtf(var + 1e-5f);
  }
  __syncthreads();
  float mean = bc[0], rstd = bc[1];
  __bf16* dst = tok + ((size_t)s * MP + b) * E_;
  dst[tid]       = (__bf16)((x0 - mean) * rstd * g[tid]       + bb[tid]);
  dst[tid + 256] = (__bf16)((x1 - mean) * rstd * g[tid + 256] + bb[tid + 256]);
}

// zero pad rows of tok, zero ctx (fp32 + bf16), reset barrier
__global__ __launch_bounds__(256) void init_kernel(__bf16* tok, float* ctxf, __bf16* ctxb,
                                                   unsigned* bar) {
  size_t i = (size_t)blockIdx.x * blockDim.x + threadIdx.x;
  size_t stride = (size_t)gridDim.x * blockDim.x;
  for (size_t j = i; j < (size_t)S_ * 8 * E_; j += stride) {
    int s = (int)(j / (8 * E_));
    int rem = (int)(j % (8 * E_));
    int r = 8 + rem / E_;
    int e = rem % E_;
    tok[((size_t)s * MP + r) * E_ + e] = (__bf16)0.f;
  }
  for (size_t j = i; j < (size_t)MP * C_; j += stride) {
    ctxf[j] = 0.f;
    ctxb[j] = (__bf16)0.f;
  }
  if (i < 2) bar[i] = 0u;
}

// ---------------- phase 2: persistent recurrence (64 blocks, grid barrier) ----------------
constexpr int NB = 64;

__global__ __launch_bounds__(256, 1) void recurrence_kernel(
    const __bf16* __restrict__ tok, float* ctxf, __bf16* ctxb,
    __bf16* h1b, __bf16* h2b, __bf16* h2all, float* gateb, float* candb,
    const __bf16* __restrict__ W1t, const __bf16* __restrict__ W2t,
    const __bf16* __restrict__ Wgt, const __bf16* __restrict__ Wct,
    const float* __restrict__ b1, const float* __restrict__ b2,
    const float* __restrict__ bg, const float* __restrict__ bcv,
    const float* __restrict__ cng, const float* __restrict__ cnb,
    float* __restrict__ traj, unsigned* bar) {
  const int blk = blockIdx.x;
  const int tid = threadIdx.x, wave = tid >> 5, lane = tid & 31;
  __shared__ float red[8][256];
  unsigned lsense = 0;

  const int r_ = tid >> 5, l_ = tid & 31;
  const int m_out = r_ + ((l_ & 16) ? 8 : 0);
  const int n_out = l_ & 15;

  for (int t = 0; t < S_; ++t) {
    const __bf16* tok_t = tok + (size_t)t * MP * E_;

    // ---- h1 = relu(fnn_in @ W1 + b1)
    {
      int n0 = blk * 16;
      v8f acc = {};
      int kbase = wave * 128;
      #pragma unroll
      for (int kk = 0; kk < 4; ++kk) {
        int k0 = kbase + kk * 32;
        v16bf a = load_fnn_frag(tok_t, ctxb, lane, k0);
        v16bf b = load_rm_frag(W1t, EC_, n0, k0, lane);
        acc = wmma_bf16(a, b, acc);
      }
      float s = block_reduce_tile(red, acc, wave, lane, tid);
      int col = n0 + n_out;
      float v = s + b1[col];
      h1b[m_out * H_ + col] = (__bf16)(v > 0.f ? v : 0.f);
    }
    grid_barrier(bar, bar + 1, NB, lsense);

    // ---- h2 = relu(h1 @ W2 + b2)
    {
      int n0 = blk * 16;
      v8f acc = {};
      int kbase = wave * 128;
      #pragma unroll
      for (int kk = 0; kk < 4; ++kk) {
        int k0 = kbase + kk * 32;
        v16bf a = load_rm_frag(h1b, H_, 0, k0, lane);
        v16bf b = load_rm_frag(W2t, H_, n0, k0, lane);
        acc = wmma_bf16(a, b, acc);
      }
      float s = block_reduce_tile(red, acc, wave, lane, tid);
      int col = n0 + n_out;
      float v = s + b2[col];
      __bf16 hv = (__bf16)(v > 0.f ? v : 0.f);
      h2b[m_out * H_ + col] = hv;
      h2all[((size_t)t * MP + m_out) * H_ + col] = hv;
    }
    grid_barrier(bar, bar + 1, NB, lsense);

    // ---- gate/cand = sig/tanh(comb @ W{g,c} + b)
    {
      bool isGate = blk < 32;
      int n0 = (isGate ? blk : blk - 32) * 16;
      const __bf16* Wt   = isGate ? Wgt : Wct;
      const float*  bias = isGate ? bg  : bcv;
      v8f acc = {};
      int kbase = wave * 192;
      #pragma unroll
      for (int kk = 0; kk < 6; ++kk) {
        int k0 = kbase + kk * 32;
        v16bf a = load_comb_frag(h2b, ctxb, lane, k0);
        v16bf b = load_rm_frag(Wt, HC_, n0, k0, lane);
        acc = wmma_bf16(a, b, acc);
      }
      float s = block_reduce_tile(red, acc, wave, lane, tid);
      int col = n0 + n_out;
      float v = s + bias[col];
      float o = isGate ? (1.f / (1.f + __expf(-v))) : tanhf(v);
      (isGate ? gateb : candb)[m_out * C_ + col] = o;
    }
    grid_barrier(bar, bar + 1, NB, lsense);

    // ---- ctx update + LN + clip (+ traj)
    if (blk == 0) {
      int m = wave;
      float vals[16], sum = 0.f, sq = 0.f;
      #pragma unroll
      for (int i = 0; i < 16; ++i) {
        int c = lane + 32 * i;
        float g  = gateb[m * C_ + c];
        float cd = candb[m * C_ + c];
        float nv = g * cd + (1.f - g) * ctxf[m * C_ + c];
        vals[i] = nv; sum += nv; sq += nv * nv;
      }
      #pragma unroll
      for (int off = 16; off > 0; off >>= 1) {
        sum += __shfl_xor(sum, off, 32);
        sq  += __shfl_xor(sq,  off, 32);
      }
      float mean = sum * (1.f / C_);
      float var  = sq * (1.f / C_) - mean * mean;
      float rstd = rsqrtf(var + 1e-5f);
      #pragma unroll
      for (int i = 0; i < 16; ++i) {
        int c = lane + 32 * i;
        float nv = (vals[i] - mean) * rstd * cng[c] + cnb[c];
        nv = fminf(10.f, fmaxf(-10.f, nv));
        ctxf[m * C_ + c] = nv;
        ctxb[m * C_ + c] = (__bf16)nv;
        traj[((size_t)m * S_ + t) * C_ + c] = nv;
      }
    }
    grid_barrier(bar, bar + 1, NB, lsense);
  }
}

// ---------------- phase 3: logits = h2_all @ Wout + bout ----------------
// grid (V/128, S/TG). TDM stages TG timestep A-tiles (64x1024 bf16, contiguous
// in global) into padded LDS; each B fragment load feeds TG WMMAs.
__global__ __launch_bounds__(256) void logits_kernel(const __bf16* __restrict__ h2all,
                                                     const __bf16* __restrict__ Woutt,
                                                     const float* __restrict__ bout,
                                                     float* __restrict__ logits) {
  __shared__ __align__(128) char As[TG * MP * LROW];   // 130KB of 320KB/WGP LDS
  int t0   = blockIdx.y * TG;
  int wave = threadIdx.x >> 5, lane = threadIdx.x & 31;
  int n0   = blockIdx.x * 128 + wave * 16;

  if (wave == 0) {
    tdm_load_tile(h2all + (size_t)t0 * MP * H_, (unsigned)(uintptr_t)As, TG * MP);
    __builtin_amdgcn_s_wait_tensorcnt(0);
  }
  __syncthreads();

  v8f acc[TG] = {};
  for (int k0 = 0; k0 < H_; k0 += 32) {
    const __bf16* bp = Woutt + (size_t)(n0 + (lane & 15)) * H_ + k0 + ((lane & 16) ? 8 : 0);
    __builtin_prefetch(bp + 64, 0, 1);                 // global_prefetch_b8
    v16bf b = make_frag(bp, bp + 16);
    #pragma unroll
    for (int g = 0; g < TG; ++g) {
      v16bf a = load_lds_a_frag(As, g * MP, lane, k0);
      acc[g] = wmma_bf16(a, b, acc[g]);
    }
  }
  if (lane < 16) {
    int v = n0 + lane;
    float bv = bout[v];
    #pragma unroll
    for (int g = 0; g < TG; ++g)
      #pragma unroll
      for (int r = 0; r < 8; ++r)                      // lanes<16 hold batch rows m=r
        logits[((size_t)r * S_ + (t0 + g)) * V_ + v] = acc[g][r] + bv;
  }
}

// ---------------- host launcher ----------------

extern "C" void kernel_launch(void* const* d_in, const int* in_sizes, int n_in,
                              void* d_out, int out_size, void* d_ws, size_t ws_size,
                              hipStream_t stream) {
  const int*   ids  = (const int*)d_in[0];
  const float* emb  = (const float*)d_in[1];
  const float* embg = (const float*)d_in[2];
  const float* embb = (const float*)d_in[3];
  const float* W1   = (const float*)d_in[4];
  const float* b1   = (const float*)d_in[5];
  const float* W2   = (const float*)d_in[6];
  const float* b2   = (const float*)d_in[7];
  const float* Wout = (const float*)d_in[8];
  const float* bout = (const float*)d_in[9];
  const float* Wg   = (const float*)d_in[10];
  const float* bg   = (const float*)d_in[11];
  const float* Wc   = (const float*)d_in[12];
  const float* bc   = (const float*)d_in[13];
  const float* cng  = (const float*)d_in[14];
  const float* cnb  = (const float*)d_in[15];

  float* logits = (float*)d_out;
  float* traj   = (float*)d_out + (size_t)B_ * S_ * V_;

  char* ws = (char*)d_ws;
  size_t off = 0;
  auto alloc = [&](size_t bytes) { char* p = ws + off; off += (bytes + 255) & ~size_t(255); return p; };
  __bf16* Woutt = (__bf16*)alloc((size_t)V_ * H_ * 2);
  __bf16* W1t   = (__bf16*)alloc((size_t)H_ * EC_ * 2);
  __bf16* W2t   = (__bf16*)alloc((size_t)H_ * H_ * 2);
  __bf16* Wgt   = (__bf16*)alloc((size_t)C_ * HC_ * 2);
  __bf16* Wct   = (__bf16*)alloc((size_t)C_ * HC_ * 2);
  __bf16* tok   = (__bf16*)alloc((size_t)S_ * MP * E_ * 2);
  __bf16* h2all = (__bf16*)alloc((size_t)S_ * MP * H_ * 2);
  float*  ctxf  = (float*)alloc((size_t)MP * C_ * 4);
  __bf16* ctxb  = (__bf16*)alloc((size_t)MP * C_ * 2);
  __bf16* h1b   = (__bf16*)alloc((size_t)MP * H_ * 2);
  __bf16* h2b   = (__bf16*)alloc((size_t)MP * H_ * 2);
  float*  gateb = (float*)alloc((size_t)MP * C_ * 4);
  float*  candb = (float*)alloc((size_t)MP * C_ * 4);
  unsigned* bar = (unsigned*)alloc(256);

  // phase 0: weight conversion (bf16, transposed to [N][K])
  wt_transpose_kernel<<<4096, 256, 0, stream>>>(W1,   W1t,   EC_, H_);
  wt_transpose_kernel<<<4096, 256, 0, stream>>>(W2,   W2t,   H_,  H_);
  wt_transpose_kernel<<<3072, 256, 0, stream>>>(Wg,   Wgt,   HC_, C_);
  wt_transpose_kernel<<<3072, 256, 0, stream>>>(Wc,   Wct,   HC_, C_);
  wt_transpose_kernel<<<32768, 256, 0, stream>>>(Wout, Woutt, H_, V_);

  // phase 1: embedding + LN, zero pads / ctx / barrier
  embed_ln_kernel<<<B_ * S_, 256, 0, stream>>>(ids, emb, embg, embb, tok);
  init_kernel<<<4096, 256, 0, stream>>>(tok, ctxf, ctxb, bar);

  // phase 2: serial recurrence (persistent, grid barrier)
  recurrence_kernel<<<NB, 256, 0, stream>>>(tok, ctxf, ctxb, h1b, h2b, h2all,
                                            gateb, candb, W1t, W2t, Wgt, Wct,
                                            b1, b2, bg, bc, cng, cnb, traj, bar);

  // phase 3: logits GEMM (TDM-staged A, 4x B reuse)
  logits_kernel<<<dim3(V_ / 128, S_ / TG), 256, 0, stream>>>(h2all, Woutt, bout, logits);
}
